// GptOssModel_76656576299585
// MI455X (gfx1250) — matrix operand order repaired and checked by reading, not compiled, not run on hardware
//
#include <hip/hip_runtime.h>
#include <hip/hip_bf16.h>
#include <math.h>

// ---------------- problem constants ----------------
constexpr int S  = 2048;
constexpr int D  = 2048;
constexpr int H  = 16;
constexpr int KH = 4;
constexpr int HD = 128;
constexpr int WIN = 256;
constexpr int E  = 8;
constexpr int F  = 1024;
constexpr float SWIGLU_ALPHA = 1.702f;
constexpr float SWIGLU_LIMIT = 7.0f;

constexpr int SLOTS      = 4224;      // 2*S + per-expert pad-to-16 headroom
constexpr int SLOT_TILES = SLOTS / 16;

constexpr int LDS_STRIDE = 36;        // 32 data floats + 4 pad: 16B aligned, bank-conflict-free

typedef __bf16 bf16_t;
typedef __attribute__((ext_vector_type(16))) __bf16 v16bf;
typedef __attribute__((ext_vector_type(8)))  float  v8f;
typedef __attribute__((ext_vector_type(4)))  float  f32x4;
typedef int gv4i __attribute__((vector_size(16)));   // builtin's pointer element type

#if defined(__has_builtin)
#if __has_builtin(__builtin_amdgcn_global_load_async_to_lds_b128)
#define HAVE_ASYNC 1
#endif
#endif
#ifndef HAVE_ASYNC
#define HAVE_ASYNC 0
#endif

// ---------------- raw-load / convert helpers ----------------
__device__ __forceinline__ void load16_raw(const float* __restrict__ p, f32x4* r) {
  const f32x4* v = (const f32x4*)p;
  r[0] = v[0]; r[1] = v[1]; r[2] = v[2]; r[3] = v[3];
}

// A-frag raw: elems 0..15 from base+16*hf (contiguous)
__device__ __forceinline__ void load_a_raw(const float* __restrict__ base, int hf, f32x4* r) {
  const f32x4* v0 = (const f32x4*)(base + 16 * hf);
  const f32x4* v1 = (const f32x4*)(base + 16 * hf + 8);
  r[0] = v0[0]; r[1] = v0[1]; r[2] = v1[0]; r[3] = v1[1];
}

__device__ __forceinline__ v16bf cvt_frag(const f32x4* r) {
  v16bf f;
#pragma unroll
  for (int q = 0; q < 4; ++q)
#pragma unroll
    for (int i = 0; i < 4; ++i) f[q * 4 + i] = (bf16_t)r[q][i];
  return f;
}

__device__ __forceinline__ v16bf cvt_frag16(const float* r) {
  v16bf f;
#pragma unroll
  for (int j = 0; j < 16; ++j) f[j] = (bf16_t)r[j];
  return f;
}

__device__ __forceinline__ v8f wmma_bf16(v16bf a, v16bf b, v8f c) {
  return __builtin_amdgcn_wmma_f32_16x16x32_bf16(false, a, false, b, (short)0, c, false, false);
}

// ---------------- init / bookkeeping ----------------
__global__ void init_kernel(int* cnt, int* cursor, int* slot_token) {
  int i = blockIdx.x * 256 + threadIdx.x;
  if (i < E) { cnt[i] = 0; cursor[i] = 0; }
  if (i < SLOTS) slot_token[i] = -1;
}

// ---------------- RMSNorm ----------------
__global__ void rmsnorm_kernel(const float* __restrict__ x, const float* __restrict__ w,
                               float* __restrict__ out) {
  int n = blockIdx.x;
  const float* xr = x + (size_t)n * D;
  float s = 0.f;
  for (int d = threadIdx.x; d < D; d += 256) { float v = xr[d]; s += v * v; }
  __shared__ float red[256];
  red[threadIdx.x] = s;
  __syncthreads();
  for (int off = 128; off > 0; off >>= 1) {
    if (threadIdx.x < off) red[threadIdx.x] += red[threadIdx.x + off];
    __syncthreads();
  }
  float inv = rsqrtf(red[0] / (float)D + 1e-5f);
  for (int d = threadIdx.x; d < D; d += 256) out[(size_t)n * D + d] = xr[d] * inv * w[d];
}

// ---------------- tiled GEMM: C = A @ W^T + bias (+resid) ----------------
// Block = 128 threads (4 waves). Block tile 64(M) x 64(N), k-step 32.
// A(64x32) and B(64x32) fp32 tiles staged in LDS (async double-buffered when
// available), shared by all 4 waves; each wave computes 16(M) x 64(N).
__global__ __launch_bounds__(128) void gemm_at_kernel(
    const float* __restrict__ A, const float* __restrict__ W,
    const float* __restrict__ bias, const float* __restrict__ resid,
    float* __restrict__ C, int M, int N, int Kd) {
  __shared__ float lsA[2][64 * LDS_STRIDE];
  __shared__ float lsB[2][64 * LDS_STRIDE];

  int tid = threadIdx.x;
  int w = tid >> 5;
  int lane = tid & 31;
  int hf = lane >> 4, r = lane & 15;
  int tn = blockIdx.x, tm = blockIdx.y;

  // staging assignment: thread -> (row, half of 32-float k-chunk)
  int srow = tid >> 1, shp = tid & 1;
  const float* gA = A + (size_t)(tm * 64 + srow) * Kd + shp * 16;
  const float* gB = W + (size_t)(tn * 64 + srow) * Kd + shp * 16;
  int loff = srow * LDS_STRIDE + shp * 16;

  v8f zero = {};
  v8f acc[4];
#pragma unroll
  for (int t = 0; t < 4; ++t) acc[t] = zero;

#if HAVE_ASYNC
#define STAGE_ASYNC(g, l)                                                                \
  do {                                                                                   \
    __builtin_amdgcn_global_load_async_to_lds_b128((gv4i*)(g), (gv4i*)(l), 0, 0);        \
    __builtin_amdgcn_global_load_async_to_lds_b128((gv4i*)(g), (gv4i*)(l), 16, 0);       \
    __builtin_amdgcn_global_load_async_to_lds_b128((gv4i*)(g), (gv4i*)(l), 32, 0);       \
    __builtin_amdgcn_global_load_async_to_lds_b128((gv4i*)(g), (gv4i*)(l), 48, 0);       \
  } while (0)
#else
#define STAGE_ASYNC(g, l) do {} while (0)
#endif

#define STAGE_TILE(buf, kk)                                                              \
  do {                                                                                   \
    const float* ga = gA + (kk);                                                         \
    const float* gb = gB + (kk);                                                         \
    float* la = &lsA[buf][loff];                                                         \
    float* lb = &lsB[buf][loff];                                                         \
    if (HAVE_ASYNC) {                                                                    \
      STAGE_ASYNC(ga, la);                                                               \
      STAGE_ASYNC(gb, lb);                                                               \
    } else {                                                                             \
      f32x4 ra[4], rb[4];                                                                \
      load16_raw(ga, ra);                                                                \
      load16_raw(gb, rb);                                                                \
      ((f32x4*)la)[0] = ra[0]; ((f32x4*)la)[1] = ra[1];                                  \
      ((f32x4*)la)[2] = ra[2]; ((f32x4*)la)[3] = ra[3];                                  \
      ((f32x4*)lb)[0] = rb[0]; ((f32x4*)lb)[1] = rb[1];                                  \
      ((f32x4*)lb)[2] = rb[2]; ((f32x4*)lb)[3] = rb[3];                                  \
    }                                                                                    \
  } while (0)

  int cur = 0;
  STAGE_TILE(0, 0);
  for (int k0 = 0; k0 < Kd; k0 += 32) {
    bool more = (k0 + 32) < Kd;
    if (more) STAGE_TILE(cur ^ 1, k0 + 32);
#if HAVE_ASYNC
    // ASYNCcnt completes in order: <=8 outstanding => current tile's 8 loads done
    if (more) asm volatile("s_wait_asynccnt 0x8" ::: "memory");
    else      asm volatile("s_wait_asynccnt 0x0" ::: "memory");
#endif
    __syncthreads();
    // compute from LDS tile `cur`
    const float* At = &lsA[cur][0];
    const float* Bt = &lsB[cur][0];
    f32x4 araw[4];
    load_a_raw(At + (w * 16 + r) * LDS_STRIDE, hf, araw);
    v16bf a = cvt_frag(araw);
#pragma unroll
    for (int t = 0; t < 4; ++t) {
      f32x4 braw[4];
      load16_raw(Bt + (t * 16 + r) * LDS_STRIDE + 16 * hf, braw);
      v16bf b = cvt_frag(braw);
      acc[t] = wmma_bf16(a, b, acc[t]);
    }
    __syncthreads();  // all waves done reading before this buffer is overwritten
    cur ^= 1;
  }
#undef STAGE_TILE
#undef STAGE_ASYNC

#pragma unroll
  for (int t = 0; t < 4; ++t) {
    int col = tn * 64 + t * 16 + r;
    float bv = bias ? bias[col] : 0.f;
#pragma unroll
    for (int g = 0; g < 8; ++g) {
      int row = tm * 64 + w * 16 + g + 8 * hf;
      size_t idx = (size_t)row * N + col;
      float v = acc[t][g] + bv;
      if (resid) v += resid[idx];
      C[idx] = v;
    }
  }
}

// ---------------- RoPE (in-place on q and k) ----------------
__global__ void rope_kernel(float* __restrict__ q, float* __restrict__ k,
                            const float* __restrict__ cosb, const float* __restrict__ sinb) {
  int s = blockIdx.x;
  const float* cr = cosb + (size_t)s * HD;
  const float* sr = sinb + (size_t)s * HD;
  for (int w = threadIdx.x; w < (H + KH) * 64; w += 256) {
    int head = w >> 6, d = w & 63;
    float c = cr[d], si = sr[d];
    float* base = (head < H) ? (q + ((size_t)s * H + head) * HD)
                             : (k + ((size_t)s * KH + (head - H)) * HD);
    float lo = base[d], hi = base[d + 64];
    base[d]      = lo * c - hi * si;
    base[d + 64] = hi * c + lo * si;
  }
}

// ---------------- flash attention, sliding window + sink ----------------
__global__ __launch_bounds__(32) void attn_kernel(
    const float* __restrict__ q, const float* __restrict__ k, const float* __restrict__ v,
    const float* __restrict__ sinks, float* __restrict__ attn) {
  int lane = threadIdx.x & 31;
  int hf = lane >> 4, r = lane & 15;
  int qb = blockIdx.x * 16;
  int h = blockIdx.y;
  int kvh = h >> 2;  // H/KH = 4
  const float scale = 0.0883883476483184f;  // 1/sqrt(128)

  const float* qrow = q + ((size_t)(qb + r) * H + h) * HD;
  v16bf qf[4];
#pragma unroll
  for (int c = 0; c < 4; ++c) {
    f32x4 qraw[4];
    load_a_raw(qrow + c * 32, hf, qraw);
#pragma unroll
    for (int qq = 0; qq < 4; ++qq)
#pragma unroll
      for (int i = 0; i < 4; ++i) qf[c][qq * 4 + i] = (bf16_t)(qraw[qq][i] * scale);
  }

  float snk = sinks[h];
  float mrun[8], lrun[8];
#pragma unroll
  for (int g = 0; g < 8; ++g) { mrun[g] = snk; lrun[g] = 1.0f; }  // sink: exp(snk-m0)=1
  v8f zero = {};
  v8f acc[8];
#pragma unroll
  for (int dt = 0; dt < 8; ++dt) acc[dt] = zero;

  __shared__ float pl[16 * 32];

  int kstart = qb - (WIN - 1);
  if (kstart < 0) kstart = 0;
  kstart &= ~31;

  for (int kb = kstart; kb <= qb + 15; kb += 32) {
    v8f s2[2];
#pragma unroll
    for (int t = 0; t < 2; ++t) {
      int key = kb + t * 16 + r;
      int kc = key < S ? key : S - 1;
      const float* krow = k + ((size_t)kc * KH + kvh) * HD;
      f32x4 kraw[4][4];
#pragma unroll
      for (int c = 0; c < 4; ++c) load16_raw(krow + c * 32 + 16 * hf, kraw[c]);
      v8f sc = zero;
#pragma unroll
      for (int c = 0; c < 4; ++c) {
        v16bf bfr = cvt_frag(kraw[c]);
        sc = wmma_bf16(qf[c], bfr, sc);
      }
      s2[t] = sc;
    }
#pragma unroll
    for (int g = 0; g < 8; ++g) {
      int qi = qb + g + 8 * hf;
#pragma unroll
      for (int t = 0; t < 2; ++t) {
        int ki = kb + t * 16 + r;
        bool valid = (ki <= qi) && (qi - ki < WIN) && (ki < S);
        s2[t][g] = valid ? s2[t][g] : -1e30f;
      }
      float rm = fmaxf(s2[0][g], s2[1][g]);
#pragma unroll
      for (int off = 1; off < 16; off <<= 1) rm = fmaxf(rm, __shfl_xor(rm, off, 16));
      float mnew = fmaxf(mrun[g], rm);
      float corr = __expf(mrun[g] - mnew);
      mrun[g] = mnew;
      float p0 = __expf(s2[0][g] - mnew);
      float p1 = __expf(s2[1][g] - mnew);
      s2[0][g] = p0; s2[1][g] = p1;
      float rs = p0 + p1;
#pragma unroll
      for (int off = 1; off < 16; off <<= 1) rs += __shfl_xor(rs, off, 16);
      lrun[g] = lrun[g] * corr + rs;
#pragma unroll
      for (int dt = 0; dt < 8; ++dt) acc[dt][g] *= corr;
    }
#pragma unroll
    for (int g = 0; g < 8; ++g) {
      int row = g + 8 * hf;
      pl[row * 32 + r]      = s2[0][g];
      pl[row * 32 + 16 + r] = s2[1][g];
    }
    __syncthreads();
    v16bf pa;
    {
      const float* p0 = &pl[r * 32 + 16 * hf];
      const float* p1 = &pl[r * 32 + 16 * hf + 8];
#pragma unroll
      for (int i = 0; i < 8; ++i) { pa[i] = (bf16_t)p0[i]; pa[i + 8] = (bf16_t)p1[i]; }
    }
    __syncthreads();
#pragma unroll
    for (int dt = 0; dt < 8; ++dt) {
      float vraw[16];
      int dcol = dt * 16 + r;
#pragma unroll
      for (int j = 0; j < 16; ++j) {
        int key = kb + 16 * hf + j;
        int kc = key < S ? key : S - 1;
        vraw[j] = v[((size_t)kc * KH + kvh) * HD + dcol];
      }
      v16bf vb = cvt_frag16(vraw);
      acc[dt] = wmma_bf16(pa, vb, acc[dt]);
    }
  }
#pragma unroll
  for (int dt = 0; dt < 8; ++dt) {
#pragma unroll
    for (int g = 0; g < 8; ++g) {
      int row = qb + g + 8 * hf;
      attn[((size_t)row * H + h) * HD + dt * 16 + r] = acc[dt][g] / lrun[g];
    }
  }
}

// ---------------- router ----------------
__global__ void router_kernel(const float* __restrict__ h2, const float* __restrict__ rw,
                              const float* __restrict__ rb, int* __restrict__ topE,
                              float* __restrict__ wts, int* __restrict__ cnt) {
  int n = blockIdx.x;
  int tid = threadIdx.x;
  int e = tid & 7, part = tid >> 3;
  const float* x = h2 + (size_t)n * D;
  const float* w = rw + (size_t)e * D;
  float s = 0.f;
  int d0 = part * 64;
  for (int d = d0; d < d0 + 64; ++d) s += x[d] * w[d];
  __shared__ float red[256];
  red[tid] = s;
  __syncthreads();
  for (int off = 128; off >= 8; off >>= 1) {
    if (tid < off) red[tid] += red[tid + off];
    __syncthreads();
  }
  if (tid == 0) {
    float lg[E];
    for (int i = 0; i < E; ++i) lg[i] = red[i] + rb[i];
    float v0 = -1e30f, v1 = -1e30f; int i0 = 0, i1 = 0;
    for (int i = 0; i < E; ++i) {
      float v = lg[i];
      if (v > v0) { v1 = v0; i1 = i0; v0 = v; i0 = i; }
      else if (v > v1) { v1 = v; i1 = i; }
    }
    float w0 = 1.0f / (1.0f + __expf(v1 - v0));
    topE[n * 2] = i0; topE[n * 2 + 1] = i1;
    wts[n * 2] = w0;  wts[n * 2 + 1] = 1.0f - w0;
    atomicAdd(&cnt[i0], 1);
    atomicAdd(&cnt[i1], 1);
  }
}

__global__ void offsets_kernel(const int* __restrict__ cnt, int* __restrict__ padOff,
                               int* __restrict__ cursor) {
  if (threadIdx.x == 0) {
    int o = 0;
    for (int e = 0; e < E; ++e) { padOff[e] = o; o += (cnt[e] + 15) & ~15; }
    padOff[E] = o;
  }
  if (threadIdx.x < E) cursor[threadIdx.x] = 0;
}

__global__ void scatter_kernel(const int* __restrict__ topE, const int* __restrict__ padOff,
                               int* __restrict__ cursor, int* __restrict__ slot_token,
                               int* __restrict__ token_slot) {
  int n = blockIdx.x * 256 + threadIdx.x;
  if (n >= S) return;
  for (int j = 0; j < 2; ++j) {
    int e = topE[n * 2 + j];
    int pos = atomicAdd(&cursor[e], 1);
    int slot = padOff[e] + pos;
    slot_token[slot] = n;
    token_slot[n * 2 + j] = slot;
  }
}

// ---------------- MoE gate/up GEMMs + SwiGLU, gathered rows ----------------
__global__ __launch_bounds__(32) void moe_gateup_kernel(
    const float* __restrict__ h2, const float* __restrict__ gate_w,
    const float* __restrict__ gate_b, const float* __restrict__ up_w,
    const float* __restrict__ up_b, const int* __restrict__ padOff,
    const int* __restrict__ slot_token, float* __restrict__ act) {
  int lane = threadIdx.x & 31;
  int hf = lane >> 4, r = lane & 15;
  int ft = blockIdx.x, tile = blockIdx.y;
  int rowBase = tile * 16;
  int e = 0;
#pragma unroll
  for (int i = 1; i < E; ++i) if (rowBase >= padOff[i]) e = i;
  int tok = slot_token[rowBase + r];
  if (tok < 0) tok = 0;
  const float* arow = h2 + (size_t)tok * D;
  const float* gw = gate_w + (size_t)e * D * F;
  const float* uw = up_w + (size_t)e * D * F;
  int f0 = ft * 32 + r;
  int f1 = f0 + 16;
  v8f ga0 = {}, ga1 = {}, ua0 = {}, ua1 = {};
  for (int k0 = 0; k0 < D; k0 += 32) {
    if (k0 + 256 < D) __builtin_prefetch((const void*)(arow + k0 + 256), 0, 1);
    int kb = k0 + 16 * hf;
    f32x4 araw[4];
    load_a_raw(arow + k0, hf, araw);
    float g0r[16], g1r[16], u0r[16], u1r[16];
#pragma unroll
    for (int j = 0; j < 16; ++j) {
      const float* grow = gw + (size_t)(kb + j) * F;
      const float* urow = uw + (size_t)(kb + j) * F;
      g0r[j] = grow[f0]; g1r[j] = grow[f1];
      u0r[j] = urow[f0]; u1r[j] = urow[f1];
    }
    v16bf a = cvt_frag(araw);
    v16bf bg0 = cvt_frag16(g0r), bg1 = cvt_frag16(g1r);
    v16bf bu0 = cvt_frag16(u0r), bu1 = cvt_frag16(u1r);
    ga0 = wmma_bf16(a, bg0, ga0);
    ga1 = wmma_bf16(a, bg1, ga1);
    ua0 = wmma_bf16(a, bu0, ua0);
    ua1 = wmma_bf16(a, bu1, ua1);
  }
#pragma unroll
  for (int t = 0; t < 2; ++t) {
    int f = (t == 0) ? f0 : f1;
    const v8f& gacc = (t == 0) ? ga0 : ga1;
    const v8f& uacc = (t == 0) ? ua0 : ua1;
    float gb = gate_b[(size_t)e * F + f];
    float ub = up_b[(size_t)e * F + f];
#pragma unroll
    for (int g = 0; g < 8; ++g) {
      int row = rowBase + g + 8 * hf;
      float gate = fminf(gacc[g] + gb, SWIGLU_LIMIT);
      float up = fminf(fmaxf(uacc[g] + ub, -SWIGLU_LIMIT), SWIGLU_LIMIT);
      float glu = gate / (1.0f + __expf(-SWIGLU_ALPHA * gate));
      act[(size_t)row * F + f] = (up + 1.0f) * glu;
    }
  }
}

// ---------------- MoE down GEMM: one wave = 16 rows x 64 cols ----------------
__global__ __launch_bounds__(32) void moe_down_kernel(
    const float* __restrict__ act, const float* __restrict__ down_w,
    const float* __restrict__ down_b, const int* __restrict__ padOff,
    float* __restrict__ y) {
  int lane = threadIdx.x & 31;
  int hf = lane >> 4, r = lane & 15;
  int nt = blockIdx.x, tile = blockIdx.y;
  int rowBase = tile * 16;
  int e = 0;
#pragma unroll
  for (int i = 1; i < E; ++i) if (rowBase >= padOff[i]) e = i;
  const float* arow = act + (size_t)(rowBase + r) * F;
  const float* wbase = down_w + (size_t)e * F * D;
  v8f zero = {};
  v8f accv[4];
#pragma unroll
  for (int t = 0; t < 4; ++t) accv[t] = zero;
  for (int k0 = 0; k0 < F; k0 += 32) {
    if (k0 + 256 < F) __builtin_prefetch((const void*)(arow + k0 + 256), 0, 1);
    int kb = k0 + 16 * hf;
    f32x4 araw[4];
    load_a_raw(arow + k0, hf, araw);
    float braw[4][16];
#pragma unroll
    for (int j = 0; j < 16; ++j) {
      const float* wrow = wbase + (size_t)(kb + j) * D + nt * 64 + r;
#pragma unroll
      for (int t = 0; t < 4; ++t) braw[t][j] = wrow[t * 16];
    }
    v16bf a = cvt_frag(araw);
#pragma unroll
    for (int t = 0; t < 4; ++t) {
      v16bf b = cvt_frag16(braw[t]);
      accv[t] = wmma_bf16(a, b, accv[t]);
    }
  }
#pragma unroll
  for (int t = 0; t < 4; ++t) {
    int col = nt * 64 + t * 16 + r;
    float bb = down_b[(size_t)e * D + col];
#pragma unroll
    for (int g = 0; g < 8; ++g)
      y[(size_t)(rowBase + g + 8 * hf) * D + col] = accv[t][g] + bb;
  }
}

// ---------------- combine: out += w0*y[s0] + w1*y[s1] ----------------
__global__ void combine_kernel(const int* __restrict__ token_slot, const float* __restrict__ wts,
                               const float* __restrict__ y, float* __restrict__ out) {
  int n = blockIdx.y;
  int d = blockIdx.x * 256 + threadIdx.x;
  int s0 = token_slot[n * 2], s1 = token_slot[n * 2 + 1];
  float w0 = wts[n * 2], w1 = wts[n * 2 + 1];
  out[(size_t)n * D + d] += w0 * y[(size_t)s0 * D + d] + w1 * y[(size_t)s1 * D + d];
}

// ---------------- host launcher ----------------
extern "C" void kernel_launch(void* const* d_in, const int* in_sizes, int n_in,
                              void* d_out, int out_size, void* d_ws, size_t ws_size,
                              hipStream_t stream) {
  const float* x        = (const float*)d_in[0];
  const float* ln1_w    = (const float*)d_in[1];
  const float* ln2_w    = (const float*)d_in[2];
  const float* q_w      = (const float*)d_in[3];
  const float* q_b      = (const float*)d_in[4];
  const float* k_w      = (const float*)d_in[5];
  const float* k_b      = (const float*)d_in[6];
  const float* v_w      = (const float*)d_in[7];
  const float* v_b      = (const float*)d_in[8];
  const float* o_w      = (const float*)d_in[9];
  const float* o_b      = (const float*)d_in[10];
  const float* sinks    = (const float*)d_in[11];
  const float* router_w = (const float*)d_in[12];
  const float* router_b = (const float*)d_in[13];
  const float* gate_w   = (const float*)d_in[14];
  const float* gate_b   = (const float*)d_in[15];
  const float* up_w     = (const float*)d_in[16];
  const float* up_b     = (const float*)d_in[17];
  const float* down_w   = (const float*)d_in[18];
  const float* down_b   = (const float*)d_in[19];
  const float* cosb     = (const float*)d_in[20];
  const float* sinb     = (const float*)d_in[21];
  float* out = (float*)d_out;

  float* ws = (float*)d_ws;
  size_t off = 0;
  float* hbuf = ws + off; off += (size_t)S * D;        // h, later attn
  float* qbuf = ws + off; off += (size_t)S * H * HD;   // q, later h2
  float* kbuf = ws + off; off += (size_t)S * KH * HD;
  float* vbuf = ws + off; off += (size_t)S * KH * HD;
  float* act  = ws + off; off += (size_t)SLOTS * F;
  float* ybuf = ws + off; off += (size_t)SLOTS * D;
  float* wts  = ws + off; off += (size_t)S * 2;
  int* ip = (int*)(ws + off);
  int* cnt        = ip;            // 8
  int* cursor     = ip + 8;        // 8
  int* padOff     = ip + 16;       // 9
  int* topE       = ip + 32;       // S*2
  int* slot_token = topE + S * 2;  // SLOTS
  int* token_slot = slot_token + SLOTS;  // S*2

  init_kernel<<<(SLOTS + 255) / 256, 256, 0, stream>>>(cnt, cursor, slot_token);
  rmsnorm_kernel<<<S, 256, 0, stream>>>(x, ln1_w, hbuf);

  gemm_at_kernel<<<dim3(D / 64, S / 64), 128, 0, stream>>>(hbuf, q_w, q_b, nullptr, qbuf, S, H * HD, D);
  gemm_at_kernel<<<dim3((KH * HD) / 64, S / 64), 128, 0, stream>>>(hbuf, k_w, k_b, nullptr, kbuf, S, KH * HD, D);
  gemm_at_kernel<<<dim3((KH * HD) / 64, S / 64), 128, 0, stream>>>(hbuf, v_w, v_b, nullptr, vbuf, S, KH * HD, D);

  rope_kernel<<<S, 256, 0, stream>>>(qbuf, kbuf, cosb, sinb);
  attn_kernel<<<dim3(S / 16, H), 32, 0, stream>>>(qbuf, kbuf, vbuf, sinks, hbuf);

  // out = x + attn @ o_w.T + o_b
  gemm_at_kernel<<<dim3(D / 64, S / 64), 128, 0, stream>>>(hbuf, o_w, o_b, x, out, S, D, H * HD);

  // h2 in qbuf
  rmsnorm_kernel<<<S, 256, 0, stream>>>(out, ln2_w, qbuf);

  router_kernel<<<S, 256, 0, stream>>>(qbuf, router_w, router_b, topE, wts, cnt);
  offsets_kernel<<<1, 32, 0, stream>>>(cnt, padOff, cursor);
  scatter_kernel<<<(S + 255) / 256, 256, 0, stream>>>(topE, padOff, cursor, slot_token, token_slot);

  moe_gateup_kernel<<<dim3(F / 32, SLOT_TILES), 32, 0, stream>>>(qbuf, gate_w, gate_b, up_w, up_b,
                                                                 padOff, slot_token, act);
  moe_down_kernel<<<dim3(D / 64, SLOT_TILES), 32, 0, stream>>>(act, down_w, down_b, padOff, ybuf);
  combine_kernel<<<dim3(D / 256, S), 256, 0, stream>>>(token_slot, wts, ybuf, out);
}
